// PhiDecoderLayer_29592324670033
// MI455X (gfx1250) — compile-verified, matrix operand order checked by reading, not run on hardware
//
#include <hip/hip_runtime.h>
#include <hip/hip_bf16.h>

// ---------------------------------------------------------------------------
// Phi decoder layer for gfx1250 (MI455X): bf16 WMMA GEMMs + fused epilogues,
// with double-buffered GLOBAL_LOAD_ASYNC_TO_LDS staging (ASYNCcnt pipeline).
// ---------------------------------------------------------------------------

typedef __attribute__((ext_vector_type(16))) __bf16 v16bf;
typedef __attribute__((ext_vector_type(8)))  __bf16 v8bf;
typedef __attribute__((ext_vector_type(4)))  __bf16 v4bf;
typedef __attribute__((ext_vector_type(8)))  float  v8f;
typedef __attribute__((ext_vector_type(4)))  int    v4i;

#define EMBED 2048
#define MLPD  8192
#define SEQ   2048
#define BATCH 2

#define BM 128
#define BN 128
#define BK 32
#define GEMM_THREADS 256
// padded LDS row stride (elements): 40 elems = 80 bytes, 16B aligned
#define LDSP 40

#ifndef __has_builtin
#define __has_builtin(x) 0
#endif

#if __has_builtin(__builtin_amdgcn_global_load_async_to_lds_b128)
#define USE_ASYNC 1
typedef __attribute__((address_space(1))) v4i gv4i_t;   // global int4
typedef __attribute__((address_space(3))) v4i sv4i_t;   // LDS int4
__device__ __forceinline__ void async_cp16(const __bf16* g, __bf16* l) {
  __builtin_amdgcn_global_load_async_to_lds_b128((gv4i_t*)g, (sv4i_t*)l, 0, 0);
}
#else
#define USE_ASYNC 0
#endif

#if __has_builtin(__builtin_amdgcn_s_wait_asynccnt)
#define WAIT_ASYNC(n) __builtin_amdgcn_s_wait_asynccnt(n)
#else
#define WAIT_ASYNC(n) asm volatile("s_wait_asynccnt %0" ::"i"(n) : "memory")
#endif

// ---------------------------------------------------------------------------
// f32 -> bf16 conversion (vectorized by 4)
// ---------------------------------------------------------------------------
__global__ void cvt_f32_to_bf16(const float* __restrict__ in,
                                __bf16* __restrict__ out, long long n4) {
  long long i = (long long)blockIdx.x * blockDim.x + threadIdx.x;
  if (i >= n4) return;
  float4 f = ((const float4*)in)[i];
  v4bf o;
  o[0] = (__bf16)f.x; o[1] = (__bf16)f.y; o[2] = (__bf16)f.z; o[3] = (__bf16)f.w;
  ((v4bf*)out)[i] = o;
}

// ---------------------------------------------------------------------------
// Fragment load from LDS, honoring the CDNA5 16-bit A/B layout:
//  lanes 0-15 (group 0): K = 0..7 in VGPR0-3, K = 16..23 in VGPR4-7
//  lanes 16-31 (group 1): K = 8..15 in VGPR0-3, K = 24..31 in VGPR4-7
// ---------------------------------------------------------------------------
__device__ __forceinline__ v16bf load_frag(const __bf16* base, int lane) {
  int g = (lane >> 4) & 1;
  int m = lane & 15;
  const __bf16* p = base + m * LDSP + 8 * g;
  v8bf lo = *(const v8bf*)(p);
  v8bf hi = *(const v8bf*)(p + 16);
  v16bf f;
#pragma unroll
  for (int i = 0; i < 8; ++i) { f[i] = lo[i]; f[i + 8] = hi[i]; }
  return f;
}

// ---------------------------------------------------------------------------
// Tiled bf16 WMMA GEMM:  C[m,n] = sum_k A[m,k] * B'[k,n]   (+ fused epilogue)
//   BNN == false : B stored row-major [N][K] (NT). Double-buffered async
//                  GLOBAL_LOAD_ASYNC_TO_LDS pipeline when available.
//   BNN == true  : B stored row-major [K][N] (NN), transposed on LDS store
//                  (synchronous staging; async copies cannot transpose).
// Epilogue: val = acc*scale + bias[n] (+ resid[m,n]) (-> gelu) -> outF / outB
// blockIdx.z selects the batch via element strides.
// ---------------------------------------------------------------------------
template <bool BNN>
__global__ __launch_bounds__(GEMM_THREADS)
void gemm_wmma_bf16(const __bf16* __restrict__ A, const __bf16* __restrict__ B,
                    const float* __restrict__ bias,
                    const float* __restrict__ resid,
                    float* __restrict__ outF, __bf16* __restrict__ outB,
                    int M, int N, int K,
                    long long aStride, long long bStride, long long oStride,
                    float scale, int doGelu) {
  __shared__ __bf16 As[2][BM * LDSP];
  __shared__ __bf16 Bs[2][BN * LDSP];

  A += (long long)blockIdx.z * aStride;
  B += (long long)blockIdx.z * bStride;
  if (outF)  outF  += (long long)blockIdx.z * oStride;
  if (outB)  outB  += (long long)blockIdx.z * oStride;
  if (resid) resid += (long long)blockIdx.z * oStride;

  const int tid  = threadIdx.x;
  const int lane = tid & 31;
  const int wid  = tid >> 5;          // 8 waves
  const int waveR = wid >> 1;         // 0..3 : 32-row strip
  const int waveC = wid & 1;          // 0..1 : 64-col strip
  const int blockRow = blockIdx.y;
  const int blockCol = blockIdx.x;

  v8f acc[2][4];
#pragma unroll
  for (int i = 0; i < 2; ++i)
#pragma unroll
    for (int j = 0; j < 4; ++j) acc[i][j] = (v8f)0.0f;

  const int nk = K / BK;

  if (!BNN) {
#if USE_ASYNC
    // ---- async double-buffered pipeline ----
    const int r = tid >> 1;
    const int c = (tid & 1) * 16;
    const __bf16* Abase = A + (long long)(blockRow * BM + r) * K + c;
    const __bf16* Bbase = B + (long long)(blockCol * BN + r) * K + c;

    auto issue = [&](int buf, int kt) {
      const __bf16* ga = Abase + kt * BK;
      async_cp16(ga,     &As[buf][r * LDSP + c]);
      async_cp16(ga + 8, &As[buf][r * LDSP + c + 8]);
      const __bf16* gb = Bbase + kt * BK;
      async_cp16(gb,     &Bs[buf][r * LDSP + c]);
      async_cp16(gb + 8, &Bs[buf][r * LDSP + c + 8]);
    };

    issue(0, 0);
    for (int kt = 0; kt < nk; ++kt) {
      const int cur = kt & 1;
      const bool more = (kt + 1) < nk;
      if (more) {
        issue(cur ^ 1, kt + 1);
        WAIT_ASYNC(4);   // previous stage's 4 copies done (in-order)
      } else {
        WAIT_ASYNC(0);
      }
      __syncthreads();

      v16bf af[2], bfv[4];
#pragma unroll
      for (int i = 0; i < 2; ++i)
        af[i] = load_frag(&As[cur][(waveR * 32 + i * 16) * LDSP], lane);
#pragma unroll
      for (int j = 0; j < 4; ++j)
        bfv[j] = load_frag(&Bs[cur][(waveC * 64 + j * 16) * LDSP], lane);
#pragma unroll
      for (int i = 0; i < 2; ++i)
#pragma unroll
        for (int j = 0; j < 4; ++j)
          acc[i][j] = __builtin_amdgcn_wmma_f32_16x16x32_bf16(
              false, af[i], false, bfv[j], (short)0, acc[i][j], false, false);
      __syncthreads();  // reads done before buffer `cur` is overwritten
    }
#else
    // ---- synchronous fallback (single buffer) ----
    for (int kt = 0; kt < nk; ++kt) {
      int r = tid >> 1;
      int c = (tid & 1) * 16;
      const uint4* srcA =
          (const uint4*)(A + (long long)(blockRow * BM + r) * K + kt * BK + c);
      uint4 a0 = srcA[0];
      uint4 a1 = srcA[1];
      const uint4* srcB =
          (const uint4*)(B + (long long)(blockCol * BN + r) * K + kt * BK + c);
      uint4 b0 = srcB[0];
      uint4 b1 = srcB[1];
      *(uint4*)&As[0][r * LDSP + c]     = a0;
      *(uint4*)&As[0][r * LDSP + c + 8] = a1;
      *(uint4*)&Bs[0][r * LDSP + c]     = b0;
      *(uint4*)&Bs[0][r * LDSP + c + 8] = b1;
      __syncthreads();

      v16bf af[2], bfv[4];
#pragma unroll
      for (int i = 0; i < 2; ++i)
        af[i] = load_frag(&As[0][(waveR * 32 + i * 16) * LDSP], lane);
#pragma unroll
      for (int j = 0; j < 4; ++j)
        bfv[j] = load_frag(&Bs[0][(waveC * 64 + j * 16) * LDSP], lane);
#pragma unroll
      for (int i = 0; i < 2; ++i)
#pragma unroll
        for (int j = 0; j < 4; ++j)
          acc[i][j] = __builtin_amdgcn_wmma_f32_16x16x32_bf16(
              false, af[i], false, bfv[j], (short)0, acc[i][j], false, false);
      __syncthreads();
    }
#endif
  } else {
    // ---- NN path: transpose B tile on the LDS-store path ----
    for (int kt = 0; kt < nk; ++kt) {
      {
        int r = tid >> 1;
        int c = (tid & 1) * 16;
        const uint4* src =
            (const uint4*)(A + (long long)(blockRow * BM + r) * K + kt * BK + c);
        uint4 a0 = src[0];
        uint4 a1 = src[1];
        *(uint4*)&As[0][r * LDSP + c]     = a0;
        *(uint4*)&As[0][r * LDSP + c + 8] = a1;
      }
      {
        int k = tid >> 3;            // 0..31
        int c = (tid & 7) * 16;      // 0..112
        const __bf16* src = B + (long long)(kt * BK + k) * N + blockCol * BN + c;
        v8bf b0 = *(const v8bf*)(src);
        v8bf b1 = *(const v8bf*)(src + 8);
#pragma unroll
        for (int j = 0; j < 8; ++j) {
          Bs[0][(c + j) * LDSP + k]     = b0[j];
          Bs[0][(c + j + 8) * LDSP + k] = b1[j];
        }
      }
      __syncthreads();

      v16bf af[2], bfv[4];
#pragma unroll
      for (int i = 0; i < 2; ++i)
        af[i] = load_frag(&As[0][(waveR * 32 + i * 16) * LDSP], lane);
#pragma unroll
      for (int j = 0; j < 4; ++j)
        bfv[j] = load_frag(&Bs[0][(waveC * 64 + j * 16) * LDSP], lane);
#pragma unroll
      for (int i = 0; i < 2; ++i)
#pragma unroll
        for (int j = 0; j < 4; ++j)
          acc[i][j] = __builtin_amdgcn_wmma_f32_16x16x32_bf16(
              false, af[i], false, bfv[j], (short)0, acc[i][j], false, false);
      __syncthreads();
    }
  }

  // ---- epilogue: D layout -> lane(n), vgpr r: M = g*8 + r ----
  const int g  = lane >> 4;
  const int ln = lane & 15;
#pragma unroll
  for (int i = 0; i < 2; ++i) {
#pragma unroll
    for (int j = 0; j < 4; ++j) {
      int mBase = blockRow * BM + waveR * 32 + i * 16 + g * 8;
      int nCol  = blockCol * BN + waveC * 64 + j * 16 + ln;
      float bv = bias ? bias[nCol] : 0.0f;
#pragma unroll
      for (int r = 0; r < 8; ++r) {
        long long idx = (long long)(mBase + r) * N + nCol;
        float v = acc[i][j][r] * scale + bv;
        if (resid) v += resid[idx];
        if (doGelu) v = 0.5f * v * (1.0f + erff(v * 0.70710678118654752f));
        if (outF) outF[idx] = v;
        if (outB) outB[idx] = (__bf16)v;
      }
    }
  }
}

// ---------------------------------------------------------------------------
// Row softmax: f32 scores -> bf16 probs (row = blockIdx.x, 2048 cols)
// ---------------------------------------------------------------------------
__global__ __launch_bounds__(256)
void softmax_rows(const float* __restrict__ S, __bf16* __restrict__ P) {
  const int COLS = SEQ;
  const float* row = S + (long long)blockIdx.x * COLS;
  __bf16* prow = P + (long long)blockIdx.x * COLS;
  __shared__ float red[256];
  int tid = threadIdx.x;
  float v[8];
  float mx = -3.4e38f;
#pragma unroll
  for (int i = 0; i < 8; ++i) {
    v[i] = row[tid + i * 256];
    mx = fmaxf(mx, v[i]);
  }
  red[tid] = mx; __syncthreads();
  for (int o = 128; o > 0; o >>= 1) {
    if (tid < o) red[tid] = fmaxf(red[tid], red[tid + o]);
    __syncthreads();
  }
  mx = red[0]; __syncthreads();
  float s = 0.0f;
#pragma unroll
  for (int i = 0; i < 8; ++i) { v[i] = __expf(v[i] - mx); s += v[i]; }
  red[tid] = s; __syncthreads();
  for (int o = 128; o > 0; o >>= 1) {
    if (tid < o) red[tid] += red[tid + o];
    __syncthreads();
  }
  float inv = 1.0f / red[0];
#pragma unroll
  for (int i = 0; i < 8; ++i) prow[tid + i * 256] = (__bf16)(v[i] * inv);
}

// ---------------------------------------------------------------------------
// h = LayerNorm(x + attn) * g + b ; writes f32 h (for final residual) + bf16 h
// ---------------------------------------------------------------------------
__global__ __launch_bounds__(256)
void add_ln_rows(const float* __restrict__ x, const float* __restrict__ attn,
                 const float* __restrict__ gamma, const float* __restrict__ beta,
                 float* __restrict__ hF, __bf16* __restrict__ hB) {
  const int COLS = EMBED;
  long long base = (long long)blockIdx.x * COLS;
  __shared__ float red[256];
  int tid = threadIdx.x;
  float v[8];
  float s = 0.0f;
#pragma unroll
  for (int i = 0; i < 8; ++i) {
    int c = tid + i * 256;
    v[i] = x[base + c] + attn[base + c];
    s += v[i];
  }
  red[tid] = s; __syncthreads();
  for (int o = 128; o > 0; o >>= 1) {
    if (tid < o) red[tid] += red[tid + o];
    __syncthreads();
  }
  float mean = red[0] * (1.0f / COLS); __syncthreads();
  float s2 = 0.0f;
#pragma unroll
  for (int i = 0; i < 8; ++i) { float d = v[i] - mean; s2 += d * d; }
  red[tid] = s2; __syncthreads();
  for (int o = 128; o > 0; o >>= 1) {
    if (tid < o) red[tid] += red[tid + o];
    __syncthreads();
  }
  float rstd = rsqrtf(red[0] * (1.0f / COLS) + 1e-5f);
#pragma unroll
  for (int i = 0; i < 8; ++i) {
    int c = tid + i * 256;
    float h = (v[i] - mean) * rstd * gamma[c] + beta[c];
    hF[base + c] = h;
    hB[base + c] = (__bf16)h;
  }
}

// ---------------------------------------------------------------------------
// Host launch
// ---------------------------------------------------------------------------
extern "C" void kernel_launch(void* const* d_in, const int* in_sizes, int n_in,
                              void* d_out, int out_size, void* d_ws, size_t ws_size,
                              hipStream_t stream) {
  (void)in_sizes; (void)n_in; (void)out_size; (void)ws_size;

  const float* x    = (const float*)d_in[0];
  const float* wq   = (const float*)d_in[1];
  const float* bq   = (const float*)d_in[2];
  const float* wk   = (const float*)d_in[3];
  const float* bk   = (const float*)d_in[4];
  const float* wv   = (const float*)d_in[5];
  const float* bv   = (const float*)d_in[6];
  const float* wd   = (const float*)d_in[7];
  const float* bd   = (const float*)d_in[8];
  const float* ln_g = (const float*)d_in[9];
  const float* ln_b = (const float*)d_in[10];
  const float* w1   = (const float*)d_in[11];
  const float* b1   = (const float*)d_in[12];
  const float* w2   = (const float*)d_in[13];
  const float* b2   = (const float*)d_in[14];
  float* out = (float*)d_out;

  const long long MTOT = (long long)BATCH * SEQ;         // 4096
  const long long EE   = (long long)EMBED * EMBED;       // 4.2M
  const long long XE   = MTOT * EMBED;                   // 8.4M
  const long long ME   = (long long)MLPD * EMBED;        // 16.8M
  const long long XM   = MTOT * MLPD;                    // 33.6M
  const long long SS   = (long long)SEQ * SEQ;           // 4.2M per batch

  // ---- workspace carve-out ----
  size_t off = 0;
  auto alloc = [&](size_t bytes) -> char* {
    char* p = (char*)d_ws + off;
    off += (bytes + 255) & ~(size_t)255;
    return p;
  };
  __bf16* xb   = (__bf16*)alloc(XE * 2);
  __bf16* wqb  = (__bf16*)alloc(EE * 2);
  __bf16* wkb  = (__bf16*)alloc(EE * 2);
  __bf16* wvb  = (__bf16*)alloc(EE * 2);
  __bf16* wdb  = (__bf16*)alloc(EE * 2);
  __bf16* w1b  = (__bf16*)alloc(ME * 2);
  __bf16* w2b  = (__bf16*)alloc(ME * 2);
  __bf16* qb   = (__bf16*)alloc(XE * 2);
  __bf16* kb   = (__bf16*)alloc(XE * 2);
  __bf16* vb   = (__bf16*)alloc(XE * 2);
  float*  scF  = (float*)alloc((size_t)BATCH * SS * 4);
  __bf16* prb  = (__bf16*)alloc((size_t)BATCH * SS * 2);
  __bf16* ctxb = (__bf16*)alloc(XE * 2);
  float*  attF = (float*)alloc(XE * 4);
  float*  hF   = (float*)alloc(XE * 4);
  __bf16* hB   = (__bf16*)alloc(XE * 2);
  __bf16* m1b  = (__bf16*)alloc(XM * 2);

  // ---- 1) convert f32 inputs to bf16 ----
  auto cvt = [&](const float* src, __bf16* dst, long long n) {
    long long n4 = n / 4;
    dim3 g((unsigned)((n4 + 255) / 256));
    cvt_f32_to_bf16<<<g, 256, 0, stream>>>(src, dst, n4);
  };
  cvt(x, xb, XE);
  cvt(wq, wqb, EE); cvt(wk, wkb, EE); cvt(wv, wvb, EE); cvt(wd, wdb, EE);
  cvt(w1, w1b, ME); cvt(w2, w2b, ME);

  // ---- generic GEMM launcher ----
  auto gemm_nt = [&](const __bf16* A, const __bf16* B, const float* bias,
                     const float* resid, float* oF, __bf16* oB,
                     int M, int N, int K,
                     long long aS, long long bS, long long oS, int Z,
                     float scale, int gelu) {
    dim3 g(N / BN, M / BM, Z);
    gemm_wmma_bf16<false><<<g, GEMM_THREADS, 0, stream>>>(
        A, B, bias, resid, oF, oB, M, N, K, aS, bS, oS, scale, gelu);
  };
  auto gemm_nn = [&](const __bf16* A, const __bf16* B, float* oF, __bf16* oB,
                     int M, int N, int K,
                     long long aS, long long bS, long long oS, int Z) {
    dim3 g(N / BN, M / BM, Z);
    gemm_wmma_bf16<true><<<g, GEMM_THREADS, 0, stream>>>(
        A, B, nullptr, nullptr, oF, oB, M, N, K, aS, bS, oS, 1.0f, 0);
  };

  // ---- 2) Q/K/V projections (batches merged: M=4096) ----
  gemm_nt(xb, wqb, bq, nullptr, nullptr, qb, (int)MTOT, EMBED, EMBED, 0, 0, 0, 1, 1.0f, 0);
  gemm_nt(xb, wkb, bk, nullptr, nullptr, kb, (int)MTOT, EMBED, EMBED, 0, 0, 0, 1, 1.0f, 0);
  gemm_nt(xb, wvb, bv, nullptr, nullptr, vb, (int)MTOT, EMBED, EMBED, 0, 0, 0, 1, 1.0f, 0);

  // ---- 3) scores = Q K^T / sqrt(E)  (per batch, z=2) ----
  const float inv_sqrt_e = 0.022097086912079608f;  // 1/sqrt(2048)
  gemm_nt(qb, kb, nullptr, nullptr, scF, nullptr, SEQ, SEQ, EMBED,
          (long long)SEQ * EMBED, (long long)SEQ * EMBED, SS, BATCH,
          inv_sqrt_e, 0);

  // ---- 4) softmax -> bf16 probs ----
  softmax_rows<<<dim3((unsigned)(BATCH * SEQ)), 256, 0, stream>>>(scF, prb);

  // ---- 5) ctx = P V  (NN gemm, per batch) ----
  gemm_nn(prb, vb, nullptr, ctxb, SEQ, EMBED, SEQ,
          SS, (long long)SEQ * EMBED, (long long)SEQ * EMBED, BATCH);

  // ---- 6) attn_out = ctx Wd^T + bd ----
  gemm_nt(ctxb, wdb, bd, nullptr, attF, nullptr, (int)MTOT, EMBED, EMBED,
          0, 0, 0, 1, 1.0f, 0);

  // ---- 7) h = LN(x + attn_out) ----
  add_ln_rows<<<dim3((unsigned)(BATCH * SEQ)), 256, 0, stream>>>(
      x, attF, ln_g, ln_b, hF, hB);

  // ---- 8) mlp1 = gelu(h W1^T + b1) ----
  gemm_nt(hB, w1b, b1, nullptr, nullptr, m1b, (int)MTOT, MLPD, EMBED,
          0, 0, 0, 1, 1.0f, 1);

  // ---- 9) out = h + (mlp1 W2^T + b2)  -> d_out directly ----
  gemm_nt(m1b, w2b, b2, hF, out, nullptr, (int)MTOT, EMBED, MLPD,
          0, 0, 0, 1, 1.0f, 0);
}
